// PixelsCoordinatesAttention_69973607186637
// MI455X (gfx1250) — matrix operand order repaired and checked by reading, not compiled, not run on hardware
//
#include <hip/hip_runtime.h>
#include <hip/hip_bf16.h>

typedef __bf16 bf16_t;
typedef __attribute__((ext_vector_type(16))) __bf16 v16bf;
typedef __attribute__((ext_vector_type(8)))  __bf16 v8bf;
typedef __attribute__((ext_vector_type(8)))  float  v8f;
typedef __attribute__((ext_vector_type(4)))  int    v4i;

#define B_    4
#define N_    2048
#define PD_   512
#define CD_   64
#define ID_   512
#define H_    8
#define DH_   64
#define DQK_  128
#define BN_   (B_ * N_)          // 8192 token rows
#define SCALE_ 0.125f            // 1/sqrt(64)
#define LOG2E_ 1.4426950408889634f

#define KLD_  144                // padded LDS row stride (elements) for K tile
#define VLD_  80                 // padded LDS row stride (elements) for V tile
#define NKT_  (N_ / 64)          // 32 KV tiles

// ---------- helpers ----------

__device__ __forceinline__ bf16_t f2bf(float f) {
    unsigned u = __float_as_uint(f);
    unsigned r = u + 0x7FFFu + ((u >> 16) & 1u);   // round-to-nearest-even
    unsigned short h = (unsigned short)(r >> 16);
    bf16_t o;
    __builtin_memcpy(&o, &h, sizeof(o));
    return o;
}

// A-fragment (16xK-major rows): lanes 0-15 / 16-31 both hold M=lane&15,
// half 'hi' selects K = 16*hi + e (contiguous 16 elements).
__device__ __forceinline__ v16bf load_a_bf16(const bf16_t* __restrict__ base,
                                             int ld, int row0, int k0, int lane) {
    const int lo = lane & 15, hi = lane >> 4;
    const bf16_t* p = base + (size_t)(row0 + lo) * ld + k0;
    v8bf c0 = *reinterpret_cast<const v8bf*>(p + hi * 16);
    v8bf c1 = *reinterpret_cast<const v8bf*>(p + hi * 16 + 8);
    v16bf a;
#pragma unroll
    for (int i = 0; i < 8; ++i) { a[i] = c0[i]; a[i + 8] = c1[i]; }
    return a;
}

// B-fragment from an "N-major" matrix T (T[n][k] == B[k][n]):
// lane = N column (lane&15), K = 16*hi + e  -> one contiguous 32B load.
__device__ __forceinline__ v16bf load_b_t(const bf16_t* __restrict__ T,
                                          int ld, int n0, int k0, int lane) {
    const int lo = lane & 15, hi = lane >> 4;
    return *reinterpret_cast<const v16bf*>(T + (size_t)(n0 + lo) * ld + k0 + hi * 16);
}

__device__ __forceinline__ v8f wmma_bf16(v16bf a, v16bf b, v8f c) {
    return __builtin_amdgcn_wmma_f32_16x16x32_bf16(false, a, false, b,
                                                   (short)0, c, false, false);
}

// async global -> LDS, 16 bytes per call (ASYNCcnt-tracked)
__device__ __forceinline__ void async_cp16(const bf16_t* gsrc, bf16_t* ldst) {
    __builtin_amdgcn_global_load_async_to_lds_b128((v4i*)gsrc, (v4i*)ldst, 0, 0);
}

// ---------- prep kernels ----------

__global__ void __launch_bounds__(256) k_convert(const float* __restrict__ in,
                                                 bf16_t* __restrict__ out, int n) {
    int i = blockIdx.x * blockDim.x + threadIdx.x;
    if (i < n) out[i] = f2bf(in[i]);
}

// out[c][r] = bf16(in[r][c])   (weights: row-major (K,N) -> N-major (N,K))
__global__ void __launch_bounds__(256) k_transpose(const float* __restrict__ in,
                                                   bf16_t* __restrict__ out, int R, int C) {
    int i = blockIdx.x * blockDim.x + threadIdx.x;
    if (i < R * C) {
        int r = i / C, c = i % C;
        out[(size_t)c * R + r] = f2bf(in[i]);
    }
}

// ---------- QKV projection: pixels(8192x512) @ W_qkv(512x1536) ----------

__global__ void __launch_bounds__(256) k_gemm_qkv(const bf16_t* __restrict__ A,
                                                  const bf16_t* __restrict__ Bt,
                                                  bf16_t* __restrict__ Qcat,
                                                  bf16_t* __restrict__ Kcat,
                                                  bf16_t* __restrict__ Vt) {
    const int lane = threadIdx.x & 31, wave = threadIdx.x >> 5;
    const int bm = blockIdx.x % (BN_ / 128);
    const int bn = blockIdx.x / (BN_ / 128);
    const int row0 = bm * 128 + wave * 16;
    const int col0 = bn * 64;
    v8f acc[4] = {};
    for (int kk = 0; kk < PD_; kk += 32) {
        v16bf a = load_a_bf16(A, PD_, row0, kk, lane);
#pragma unroll
        for (int ct = 0; ct < 4; ++ct) {
            v16bf b = load_b_t(Bt, PD_, col0 + ct * 16, kk, lane);
            acc[ct] = wmma_bf16(a, b, acc[ct]);
        }
    }
    const int lo = lane & 15, hi = lane >> 4;
#pragma unroll
    for (int ct = 0; ct < 4; ++ct) {
        const int j   = col0 + ct * 16 + lo;   // 0..1535
        const int sec = j / ID_;               // 0=q 1=k 2=v
        const int h   = (j % ID_) / DH_;
        const int d   = j % DH_;
#pragma unroll
        for (int g = 0; g < 8; ++g) {
            const int r = row0 + g + 8 * hi;
            const int b_idx = r / N_, n = r % N_;
            const size_t bh = (size_t)(b_idx * H_ + h);
            float v = acc[ct][g];
            if (sec == 0) {
                Qcat[(bh * N_ + n) * DQK_ + d] = f2bf(fminf(fmaxf(v, -5.f), 5.f) * SCALE_);
            } else if (sec == 1) {
                Kcat[(bh * N_ + n) * DQK_ + d] = f2bf(fminf(fmaxf(v, -5.f), 5.f));
            } else {
                Vt[(bh * DH_ + d) * N_ + n] = f2bf(v);
            }
        }
    }
}

// ---------- coord projection: coords(8192x64) @ W_qk_c(64x1024) ----------

__global__ void __launch_bounds__(256) k_gemm_qkc(const bf16_t* __restrict__ A,
                                                  const bf16_t* __restrict__ Bt,
                                                  bf16_t* __restrict__ Qcat,
                                                  bf16_t* __restrict__ Kcat) {
    const int lane = threadIdx.x & 31, wave = threadIdx.x >> 5;
    const int bm = blockIdx.x % (BN_ / 128);
    const int bn = blockIdx.x / (BN_ / 128);
    const int row0 = bm * 128 + wave * 16;
    const int col0 = bn * 64;
    v8f acc[4] = {};
#pragma unroll
    for (int kk = 0; kk < CD_; kk += 32) {
        v16bf a = load_a_bf16(A, CD_, row0, kk, lane);
#pragma unroll
        for (int ct = 0; ct < 4; ++ct) {
            v16bf b = load_b_t(Bt, CD_, col0 + ct * 16, kk, lane);
            acc[ct] = wmma_bf16(a, b, acc[ct]);
        }
    }
    const int lo = lane & 15, hi = lane >> 4;
#pragma unroll
    for (int ct = 0; ct < 4; ++ct) {
        const int j   = col0 + ct * 16 + lo;   // 0..1023
        const int sec = j / ID_;               // 0=qc 1=kc
        const int h   = (j % ID_) / DH_;
        const int d   = j % DH_;
#pragma unroll
        for (int g = 0; g < 8; ++g) {
            const int r = row0 + g + 8 * hi;
            const int b_idx = r / N_, n = r % N_;
            const size_t bh = (size_t)(b_idx * H_ + h);
            float cv = fminf(fmaxf(acc[ct][g], -5.f), 5.f);
            if (sec == 0)
                Qcat[(bh * N_ + n) * DQK_ + DH_ + d] = f2bf(cv * SCALE_);
            else
                Kcat[(bh * N_ + n) * DQK_ + DH_ + d] = f2bf(cv);
        }
    }
}

// ---------- flash attention: per (b,h), Q(2048x128) K(2048x128) V^T(64x2048) ----------
// K/V tiles double-buffered in LDS via async global->LDS copies (ASYNCcnt).

__global__ void __launch_bounds__(256) k_flash(const bf16_t* __restrict__ Qcat,
                                               const bf16_t* __restrict__ Kcat,
                                               const bf16_t* __restrict__ Vt,
                                               bf16_t* __restrict__ attnB) {
    __shared__ alignas(64) bf16_t Klds[2][64 * KLD_];   // 2 x 18KB
    __shared__ alignas(64) bf16_t Vlds[2][DH_ * VLD_];  // 2 x 10KB
    __shared__ alignas(64) bf16_t Plds[8][16 * 64];     // per-wave P transpose, 16KB

    const int tid  = threadIdx.x;
    const int lane = tid & 31, wave = tid >> 5;
    const int lo = lane & 15, hi = lane >> 4;
    const int bh = blockIdx.x % (B_ * H_);
    const int qt = blockIdx.x / (B_ * H_);
    const int b = bh / H_, h = bh % H_;
    const bf16_t* Qh = Qcat + (size_t)bh * N_ * DQK_;
    const bf16_t* Kh = Kcat + (size_t)bh * N_ * DQK_;
    const bf16_t* Vh = Vt   + (size_t)bh * DH_ * N_;
    const int row0 = qt * 128 + wave * 16;     // query rows of this wave

    // stage one 64-key K/V tile into LDS buffer `buf` (6 async b128 per thread)
    auto stage = [&](int buf, int key) {
#pragma unroll
        for (int j = 0; j < 4; ++j) {           // K: 64 rows x 16 chunks of 16B
            int i = tid + 256 * j;
            int r = i >> 4, c = i & 15;
            async_cp16(Kh + (size_t)(key + r) * DQK_ + c * 8,
                       &Klds[buf][r * KLD_ + c * 8]);
        }
#pragma unroll
        for (int j = 0; j < 2; ++j) {           // V: 64 rows x 8 chunks of 16B
            int i = tid + 256 * j;
            int r = i >> 3, c = i & 7;
            async_cp16(Vh + (size_t)r * N_ + key + c * 8,
                       &Vlds[buf][r * VLD_ + c * 8]);
        }
    };

    // Q fragments: 16 rows x 128 features, scale pre-folded
    v16bf qf[4];
#pragma unroll
    for (int ks = 0; ks < 4; ++ks) qf[ks] = load_a_bf16(Qh, DQK_, row0, ks * 32, lane);

    v8f oacc[4] = {};
    float m[8], l[8];
#pragma unroll
    for (int g = 0; g < 8; ++g) { m[g] = -1e30f; l[g] = 0.f; }

    stage(0, 0);
    for (int t = 0; t < NKT_; ++t) {
        const int cur = t & 1;
        const int key0 = t * 64;
        if (t + 1 < NKT_) {
            stage(1 - cur, key0 + 64);
            __builtin_amdgcn_s_wait_asynccnt(6);   // only next tile's 6 outstanding
        } else {
            __builtin_amdgcn_s_wait_asynccnt(0);
        }
        __syncthreads();
        const bf16_t* Kt = &Klds[cur][0];
        const bf16_t* Vb = &Vlds[cur][0];

        // S = Q . K^T  (16 x 64), 16 WMMAs; B-frags from LDS
        v8f s[4] = {};
#pragma unroll
        for (int jt = 0; jt < 4; ++jt) {
#pragma unroll
            for (int ks = 0; ks < 4; ++ks) {
                v16bf bf = *reinterpret_cast<const v16bf*>(
                    Kt + (size_t)(jt * 16 + lo) * KLD_ + ks * 32 + hi * 16);
                s[jt] = wmma_bf16(qf[ks], bf, s[jt]);
            }
        }
        // online softmax; C-layout: row = g + 8*hi, col = jt*16 + lo
        float alpha[8];
#pragma unroll
        for (int g = 0; g < 8; ++g) {
            float mn = m[g];
#pragma unroll
            for (int jt = 0; jt < 4; ++jt) mn = fmaxf(mn, s[jt][g]);
#pragma unroll
            for (int off = 1; off < 16; off <<= 1)
                mn = fmaxf(mn, __shfl_xor(mn, off, 32));
            alpha[g] = exp2f((m[g] - mn) * LOG2E_);
            m[g] = mn;
        }
#pragma unroll
        for (int g = 0; g < 8; ++g) {
            float rs = 0.f;
#pragma unroll
            for (int jt = 0; jt < 4; ++jt) {
                float p = exp2f((s[jt][g] - m[g]) * LOG2E_);
                rs += p;
                Plds[wave][(g + 8 * hi) * 64 + jt * 16 + lo] = f2bf(p);
            }
#pragma unroll
            for (int off = 1; off < 16; off <<= 1)
                rs += __shfl_xor(rs, off, 32);
            l[g] = l[g] * alpha[g] + rs;
#pragma unroll
            for (int nt = 0; nt < 4; ++nt) oacc[nt][g] *= alpha[g];
        }
        // re-load P as A-fragments (wave-private LDS region; in-wave DS order)
        v16bf pf[2];
#pragma unroll
        for (int ks = 0; ks < 2; ++ks) {
            const bf16_t* p = &Plds[wave][lo * 64 + ks * 32];
            v8bf c0 = *reinterpret_cast<const v8bf*>(p + hi * 16);
            v8bf c1 = *reinterpret_cast<const v8bf*>(p + hi * 16 + 8);
#pragma unroll
            for (int i = 0; i < 8; ++i) { pf[ks][i] = c0[i]; pf[ks][8 + i] = c1[i]; }
        }
        // O += P . V   (B[k][n] = Vlds[n_feat][k_key]), 8 WMMAs
#pragma unroll
        for (int nt = 0; nt < 4; ++nt) {
#pragma unroll
            for (int ks = 0; ks < 2; ++ks) {
                v16bf bf = *reinterpret_cast<const v16bf*>(
                    Vb + (size_t)(nt * 16 + lo) * VLD_ + ks * 32 + hi * 16);
                oacc[nt] = wmma_bf16(pf[ks], bf, oacc[nt]);
            }
        }
        __syncthreads();   // all waves done reading before next stage overwrites
    }
    // finalize: normalize by row sums, write (b, n, h*64+d) bf16
#pragma unroll
    for (int g = 0; g < 8; ++g) {
        float inv = 1.f / l[g];
        int r = row0 + g + 8 * hi;
        bf16_t* op = attnB + ((size_t)b * N_ + r) * ID_ + h * DH_;
#pragma unroll
        for (int nt = 0; nt < 4; ++nt)
            op[nt * 16 + lo] = f2bf(oacc[nt][g] * inv);
    }
}

// ---------- output projection: attn(8192x512) @ W_out(512x512) + b ----------

__global__ void __launch_bounds__(256) k_gemm_out(const bf16_t* __restrict__ A,
                                                  const bf16_t* __restrict__ Bt,
                                                  const float* __restrict__ bias,
                                                  float* __restrict__ out) {
    const int lane = threadIdx.x & 31, wave = threadIdx.x >> 5;
    const int bm = blockIdx.x % (BN_ / 128);
    const int bn = blockIdx.x / (BN_ / 128);
    const int row0 = bm * 128 + wave * 16;
    const int col0 = bn * 64;
    v8f acc[4] = {};
    for (int kk = 0; kk < ID_; kk += 32) {
        v16bf a = load_a_bf16(A, ID_, row0, kk, lane);
#pragma unroll
        for (int ct = 0; ct < 4; ++ct) {
            v16bf b = load_b_t(Bt, ID_, col0 + ct * 16, kk, lane);
            acc[ct] = wmma_bf16(a, b, acc[ct]);
        }
    }
    const int lo = lane & 15, hi = lane >> 4;
#pragma unroll
    for (int ct = 0; ct < 4; ++ct) {
        int j = col0 + ct * 16 + lo;
        float bj = bias[j];
#pragma unroll
        for (int g = 0; g < 8; ++g) {
            int r = row0 + g + 8 * hi;
            out[(size_t)r * PD_ + j] = acc[ct][g] + bj;
        }
    }
}

// ---------- launch ----------

extern "C" void kernel_launch(void* const* d_in, const int* in_sizes, int n_in,
                              void* d_out, int out_size, void* d_ws, size_t ws_size,
                              hipStream_t stream) {
    (void)in_sizes; (void)n_in; (void)out_size; (void)ws_size;
    const float* pixels = (const float*)d_in[0];
    const float* coords = (const float*)d_in[1];
    const float* W_qkv  = (const float*)d_in[2];
    const float* W_qk_c = (const float*)d_in[3];
    const float* W_out  = (const float*)d_in[4];
    const float* b_out  = (const float*)d_in[5];

    char* ws = (char*)d_ws;
    size_t off = 0;
    bf16_t* pixB  = (bf16_t*)(ws + off); off += (size_t)BN_ * PD_ * 2;      // 8.39 MB
    bf16_t* cooB  = (bf16_t*)(ws + off); off += (size_t)BN_ * CD_ * 2;      // 1.05 MB
    bf16_t* WqkvT = (bf16_t*)(ws + off); off += (size_t)PD_ * (3 * ID_) * 2;// 1.57 MB
    bf16_t* WqkcT = (bf16_t*)(ws + off); off += (size_t)CD_ * (2 * ID_) * 2;// 0.13 MB
    bf16_t* WoutT = (bf16_t*)(ws + off); off += (size_t)ID_ * PD_ * 2;      // 0.52 MB
    bf16_t* Qcat  = (bf16_t*)(ws + off); off += (size_t)B_ * H_ * N_ * DQK_ * 2; // 16.8 MB
    bf16_t* Kcat  = (bf16_t*)(ws + off); off += (size_t)B_ * H_ * N_ * DQK_ * 2; // 16.8 MB
    bf16_t* Vtb   = (bf16_t*)(ws + off); off += (size_t)B_ * H_ * DH_ * N_ * 2;  // 8.39 MB
    bf16_t* attnB = (bf16_t*)(ws + off); off += (size_t)BN_ * ID_ * 2;      // 8.39 MB

    // prep: convert activations, transpose+convert weights
    k_convert<<<(BN_ * PD_ + 255) / 256, 256, 0, stream>>>(pixels, pixB, BN_ * PD_);
    k_convert<<<(BN_ * CD_ + 255) / 256, 256, 0, stream>>>(coords, cooB, BN_ * CD_);
    k_transpose<<<(PD_ * 3 * ID_ + 255) / 256, 256, 0, stream>>>(W_qkv, WqkvT, PD_, 3 * ID_);
    k_transpose<<<(CD_ * 2 * ID_ + 255) / 256, 256, 0, stream>>>(W_qk_c, WqkcT, CD_, 2 * ID_);
    k_transpose<<<(ID_ * PD_ + 255) / 256, 256, 0, stream>>>(W_out, WoutT, ID_, PD_);

    // projections (WMMA)
    k_gemm_qkv<<<(BN_ / 128) * (3 * ID_ / 64), 256, 0, stream>>>(pixB, WqkvT, Qcat, Kcat, Vtb);
    k_gemm_qkc<<<(BN_ / 128) * (2 * ID_ / 64), 256, 0, stream>>>(cooB, WqkcT, Qcat, Kcat);

    // fused flash attention (WMMA + async global->LDS staging + online softmax)
    k_flash<<<B_ * H_ * (N_ / 128), 256, 0, stream>>>(Qcat, Kcat, Vtb, attnB);

    // output projection + bias (WMMA)
    k_gemm_out<<<(BN_ / 128) * (PD_ / 64), 256, 0, stream>>>(attnB, WoutT, b_out, (float*)d_out);
}
